// LowrankLearnableHash_57483842290115
// MI455X (gfx1250) — matrix operand order repaired and checked by reading, not compiled, not run on hardware
//
#include <hip/hip_runtime.h>
#include <hip/hip_bf16.h>

typedef __attribute__((ext_vector_type(16))) _Float16 v16h;
typedef __attribute__((ext_vector_type(8)))  _Float16 v8h;
typedef __attribute__((ext_vector_type(8)))  float    v8f;

#define NRAYS   4096
#define NSAMP   256
#define NPTS    (NRAYS * NSAMP)      // 1,048,576
#define RADIUS_F 1.3f

// ---------------------------------------------------------------------------
// CDNA5 WMMA fragment helpers (wave32, V_WMMA_F32_16X16X32_F16 layouts)
// A (16x32 f16): lanes 0-15 hold M=0..15; g=lane>>4 selects K-halves:
//   element e -> K = kBase + (e>>3)*16 + g*8 + (e&7)
//   => per lane TWO contiguous 16-byte runs -> 2x ds_load_b128.
// B (32x16 f16): lane col n=lane&15; element e -> K = kBase + (e>>3)*16 + g*8 + (e&7).
//   Weights are staged pre-swizzled so each lane's 16 halves are contiguous.
// C/D (16x16 f32): lane col n=lane&15; VGPR v -> row M = g*8 + v.
// ---------------------------------------------------------------------------
__device__ __forceinline__ v16h load_a_frag(const _Float16* A, int row,
                                            int kBase, int g) {
  const _Float16* base = A + row * 64 + kBase + g * 8;
  v8h lo = *(const v8h*)(base);        // elements 0..7
  v8h hi = *(const v8h*)(base + 16);   // elements 8..15
  return __builtin_shufflevector(lo, hi, 0, 1, 2, 3, 4, 5, 6, 7,
                                         8, 9, 10, 11, 12, 13, 14, 15);
}

// Swizzled weight storage: tile (kt, nt) holds 32 lanes x 16 halves contiguous.
__device__ __forceinline__ v16h load_b_sw(const _Float16* Wsw, int kt, int nt,
                                          int kT, int lane) {
  return *(const v16h*)(Wsw + (((nt * kT + kt) * 32 + lane) << 4));
}

// Stage W (Kvalid x Nvalid, row stride Nstride) into fragment-order f16 LDS,
// zero-padded to kT*32 x nT*16.
__device__ __forceinline__ void stage_weights_sw(_Float16* dst, const float* W,
                                                 int Kvalid, int Nvalid,
                                                 int Nstride, int kT, int nT,
                                                 int tid) {
  int total = kT * nT * 512;
  for (int i = tid; i < total; i += 128) {
    int tile = i >> 9, within = i & 511;
    int lane = within >> 4, e = within & 15;
    int kt = tile % kT, nt = tile / kT;
    int k = kt * 32 + ((e >> 3) << 4) + ((lane >> 4) << 3) + (e & 7);
    int n = nt * 16 + (lane & 15);
    dst[i] = (_Float16)((k < Kvalid && n < Nvalid) ? W[k * Nstride + n] : 0.f);
  }
}

__device__ __forceinline__ v8f wmma32(v16h a, v16h b, v8f c) {
  return __builtin_amdgcn_wmma_f32_16x16x32_f16(false, a, false, b,
                                                (short)0, c, false, false);
}

// ===========================================================================
// Kernel 1: per-point gather + WMMA MLPs.
// 128 threads = 4 waves; each wave owns 32 points (two 16-row WMMA tiles).
// ===========================================================================
__global__ __launch_bounds__(128) void point_kernel(
    const float* __restrict__ rays_o, const float* __restrict__ rays_d,
    const float* __restrict__ plane_grid, const float* __restrict__ feat_grid,
    const float* __restrict__ W1, const float* __restrict__ b1,
    const float* __restrict__ W2, const float* __restrict__ b2,
    const float* __restrict__ Wc1, const float* __restrict__ bc1,
    const float* __restrict__ Wc2, const float* __restrict__ bc2,
    float* __restrict__ sigma_out, float* __restrict__ rgb_out)
{
  __shared__ __align__(32) _Float16 sW1[64 * 64];   // 35x64 -> 64x64, swizzled
  __shared__ __align__(32) _Float16 sW2[64 * 16];   // 64x16, swizzled
  __shared__ __align__(32) _Float16 sWc1[32 * 64];  // 18x64 -> 32x64, swizzled
  __shared__ __align__(32) _Float16 sWc2[64 * 16];  // 64x3 -> 64x16, swizzled
  __shared__ float sB1[64], sB2[16], sBc1[64], sBc2[16];
  __shared__ __align__(32) _Float16 sA[4 * 32 * 64]; // per-wave activation tile
  __shared__ float sH[4 * 32 * 16];                  // per-wave f32 head output

  const int tid  = threadIdx.x;
  const int wave = tid >> 5;
  const int lane = tid & 31;
  const int g    = lane >> 4;
  const int mrow = lane & 15;
  const int ncol = lane & 15;

  // ---- cooperative weight staging: f32 -> f16 fragment-order, zero-padded ----
  stage_weights_sw(sW1,  W1,  35, 64, 64, 2, 4, tid);
  stage_weights_sw(sW2,  W2,  64, 16, 16, 2, 1, tid);
  stage_weights_sw(sWc1, Wc1, 18, 64, 64, 1, 4, tid);
  stage_weights_sw(sWc2, Wc2, 64,  3,  3, 2, 1, tid);
  if (tid < 64) { sB1[tid] = b1[tid]; sBc1[tid] = bc1[tid]; }
  if (tid < 16) { sB2[tid] = b2[tid]; sBc2[tid] = (tid < 3) ? bc2[tid] : 0.f; }

  _Float16* Aw = sA + wave * 32 * 64;
  float*    Hw = sH + wave * 32 * 16;

  // ---- per-point ray setup + tri-plane + trilinear gather ----
  const int p   = blockIdx.x * 128 + wave * 32 + lane;
  const int ray = p >> 8;
  const int smp = p & 255;

  float ox = rays_o[ray * 3 + 0], oy = rays_o[ray * 3 + 1], oz = rays_o[ray * 3 + 2];
  float dx = rays_d[ray * 3 + 0], dy = rays_d[ray * 3 + 1], dz = rays_d[ray * 3 + 2];
  float rn = __frsqrt_rn(dx * dx + dy * dy + dz * dz);
  dx *= rn; dy *= rn; dz *= rn;
  float bb = ox * dx + oy * dy + oz * dz;
  float cc = ox * ox + oy * oy + oz * oz - RADIUS_F * RADIUS_F;
  float disc = __fsqrt_rn(fmaxf(bb * bb - cc, 0.f));
  float tn = fmaxf(-bb - disc, 0.f);
  float tf = fmaxf(-bb + disc, tn);
  float t  = tn + ((float)smp + 0.5f) * (1.f / 256.f) * (tf - tn);
  float px = (ox + t * dx) * (1.f / RADIUS_F);
  float py = (oy + t * dy) * (1.f / RADIUS_F);
  float pz = (oz + t * dz) * (1.f / RADIUS_F);

  // low-rank tri-plane: product over 3 planes per (out,rank) channel
  float pf[12];
#pragma unroll
  for (int i = 0; i < 12; ++i) pf[i] = 1.f;
  const float cu[3] = { px, px, py };
  const float cv[3] = { py, pz, pz };
#pragma unroll
  for (int pl = 0; pl < 3; ++pl) {
    float x = fminf(fmaxf((cu[pl] + 1.f) * 0.5f * 255.f, 0.f), 255.f);
    float y = fminf(fmaxf((cv[pl] + 1.f) * 0.5f * 255.f, 0.f), 255.f);
    int x0 = (int)x, y0 = (int)y;
    int x1 = min(x0 + 1, 255), y1 = min(y0 + 1, 255);
    float wx = x - (float)x0, wy = y - (float)y0;
    float w00 = (1.f - wx) * (1.f - wy), w01 = wx * (1.f - wy);
    float w10 = (1.f - wx) * wy,        w11 = wx * wy;
    const float* gp = plane_grid + pl * 12 * 65536;
#pragma unroll
    for (int ch = 0; ch < 12; ++ch) {
      const float* gc = gp + ch * 65536;
      float v = gc[y0 * 256 + x0] * w00 + gc[y0 * 256 + x1] * w01 +
                gc[y1 * 256 + x0] * w10 + gc[y1 * 256 + x1] * w11;
      pf[ch] *= v;
    }
  }
  float ix = pf[0] + pf[1] + pf[2]  + pf[3];
  float iy = pf[4] + pf[5] + pf[6]  + pf[7];
  float iz = pf[8] + pf[9] + pf[10] + pf[11];

  {
    float x = fminf(fmaxf((ix + 1.f) * 0.5f * 127.f, 0.f), 127.f);
    float y = fminf(fmaxf((iy + 1.f) * 0.5f * 127.f, 0.f), 127.f);
    float z = fminf(fmaxf((iz + 1.f) * 0.5f * 127.f, 0.f), 127.f);
    int x0 = (int)x, y0 = (int)y, z0 = (int)z;
    int x1 = min(x0 + 1, 127), y1 = min(y0 + 1, 127), z1 = min(z0 + 1, 127);
    float wx = x - (float)x0, wy = y - (float)y0, wz = z - (float)z0;
    float w000 = (1 - wz) * (1 - wy) * (1 - wx), w001 = (1 - wz) * (1 - wy) * wx;
    float w010 = (1 - wz) * wy * (1 - wx),       w011 = (1 - wz) * wy * wx;
    float w100 = wz * (1 - wy) * (1 - wx),       w101 = wz * (1 - wy) * wx;
    float w110 = wz * wy * (1 - wx),             w111 = wz * wy * wx;
    int b000 = z0 * 16384 + y0 * 128 + x0, b001 = z0 * 16384 + y0 * 128 + x1;
    int b010 = z0 * 16384 + y1 * 128 + x0, b011 = z0 * 16384 + y1 * 128 + x1;
    int b100 = z1 * 16384 + y0 * 128 + x0, b101 = z1 * 16384 + y0 * 128 + x1;
    int b110 = z1 * 16384 + y1 * 128 + x0, b111 = z1 * 16384 + y1 * 128 + x1;
    // channels grouped by 8 -> vector ds_store_b128 into the A tile
#pragma unroll
    for (int cq = 0; cq < 4; ++cq) {
      v8h chunk;
#pragma unroll
      for (int e = 0; e < 8; ++e) {
        const float* gc = feat_grid + (cq * 8 + e) * 2097152;
        float v = gc[b000] * w000 + gc[b001] * w001 + gc[b010] * w010 + gc[b011] * w011 +
                  gc[b100] * w100 + gc[b101] * w101 + gc[b110] * w110 + gc[b111] * w111;
        chunk[e] = (_Float16)v;
      }
      *(v8h*)(Aw + lane * 64 + cq * 8) = chunk;
    }
    v8h tail = {};
    tail[0] = (_Float16)dx; tail[1] = (_Float16)dy; tail[2] = (_Float16)dz;
    *(v8h*)(Aw + lane * 64 + 32) = tail;
    v8h zero8 = {};
    *(v8h*)(Aw + lane * 64 + 40) = zero8;
    *(v8h*)(Aw + lane * 64 + 48) = zero8;
    *(v8h*)(Aw + lane * 64 + 56) = zero8;
  }
  __syncthreads();

  // ---- layer 1: (32x64) @ (64x64) + b1, ReLU  -> back into Aw as f16 ----
  v16h a00 = load_a_frag(Aw, mrow,      0,  g);
  v16h a01 = load_a_frag(Aw, mrow,      32, g);
  v16h a10 = load_a_frag(Aw, 16 + mrow, 0,  g);
  v16h a11 = load_a_frag(Aw, 16 + mrow, 32, g);
  __syncthreads();  // A captured in VGPRs; safe to overwrite Aw below

#pragma unroll
  for (int rt = 0; rt < 2; ++rt) {
    v16h aK0 = rt ? a10 : a00;
    v16h aK1 = rt ? a11 : a01;
#pragma unroll
    for (int nt = 0; nt < 4; ++nt) {
      v16h bf0 = load_b_sw(sW1, 0, nt, 2, lane);
      v16h bf1 = load_b_sw(sW1, 1, nt, 2, lane);
      v8f acc = {};
      acc = wmma32(aK0, bf0, acc);
      acc = wmma32(aK1, bf1, acc);
#pragma unroll
      for (int v = 0; v < 8; ++v) {
        int row = rt * 16 + g * 8 + v;
        float hv = acc[v] + sB1[nt * 16 + ncol];
        Aw[row * 64 + nt * 16 + ncol] = (_Float16)fmaxf(hv, 0.f);
      }
    }
  }
  __syncthreads();

  // ---- layer 2: (32x64) @ (64x16) + b2 -> Hw (f32) ----
#pragma unroll
  for (int rt = 0; rt < 2; ++rt) {
    v16h aK0 = load_a_frag(Aw, rt * 16 + mrow, 0,  g);
    v16h aK1 = load_a_frag(Aw, rt * 16 + mrow, 32, g);
    v16h bf0 = load_b_sw(sW2, 0, 0, 2, lane);
    v16h bf1 = load_b_sw(sW2, 1, 0, 2, lane);
    v8f acc = {};
    acc = wmma32(aK0, bf0, acc);
    acc = wmma32(aK1, bf1, acc);
#pragma unroll
    for (int v = 0; v < 8; ++v) {
      int row = rt * 16 + g * 8 + v;
      Hw[row * 16 + ncol] = acc[v] + sB2[ncol];
    }
  }
  __syncthreads();

  // ---- sigma head + color-MLP input repack: [geo(15), dir(3), 0..] -> K=32 ----
  sigma_out[p] = __expf(Hw[lane * 16 + 0]);
  {
    v8h c0 = {}, c1 = {}, c2 = {}, c3 = {};
#pragma unroll
    for (int j = 0; j < 8; ++j) c0[j] = (_Float16)Hw[lane * 16 + 1 + j];  // geo 0..7
#pragma unroll
    for (int j = 0; j < 7; ++j) c1[j] = (_Float16)Hw[lane * 16 + 9 + j];  // geo 8..14
    c1[7] = (_Float16)dx;   // col 15
    c2[0] = (_Float16)dy;   // col 16
    c2[1] = (_Float16)dz;   // col 17
    *(v8h*)(Aw + lane * 64 + 0)  = c0;
    *(v8h*)(Aw + lane * 64 + 8)  = c1;
    *(v8h*)(Aw + lane * 64 + 16) = c2;
    *(v8h*)(Aw + lane * 64 + 24) = c3;
  }
  __syncthreads();

  // ---- color layer 1: (32x32) @ (32x64) + bc1, ReLU -> Aw ----
  v16h ca0 = load_a_frag(Aw, mrow,      0, g);
  v16h ca1 = load_a_frag(Aw, 16 + mrow, 0, g);
  __syncthreads();
#pragma unroll
  for (int rt = 0; rt < 2; ++rt) {
    v16h aK = rt ? ca1 : ca0;
#pragma unroll
    for (int nt = 0; nt < 4; ++nt) {
      v16h bf = load_b_sw(sWc1, 0, nt, 1, lane);
      v8f acc = {};
      acc = wmma32(aK, bf, acc);
#pragma unroll
      for (int v = 0; v < 8; ++v) {
        int row = rt * 16 + g * 8 + v;
        float hv = acc[v] + sBc1[nt * 16 + ncol];
        Aw[row * 64 + nt * 16 + ncol] = (_Float16)fmaxf(hv, 0.f);
      }
    }
  }
  __syncthreads();

  // ---- color layer 2: (32x64) @ (64x16) + bc2 -> Hw ----
#pragma unroll
  for (int rt = 0; rt < 2; ++rt) {
    v16h aK0 = load_a_frag(Aw, rt * 16 + mrow, 0,  g);
    v16h aK1 = load_a_frag(Aw, rt * 16 + mrow, 32, g);
    v16h bf0 = load_b_sw(sWc2, 0, 0, 2, lane);
    v16h bf1 = load_b_sw(sWc2, 1, 0, 2, lane);
    v8f acc = {};
    acc = wmma32(aK0, bf0, acc);
    acc = wmma32(aK1, bf1, acc);
#pragma unroll
    for (int v = 0; v < 8; ++v) {
      int row = rt * 16 + g * 8 + v;
      Hw[row * 16 + ncol] = acc[v] + sBc2[ncol];
    }
  }
  __syncthreads();

  // ---- sigmoid + store point rgb ----
#pragma unroll
  for (int c = 0; c < 3; ++c) {
    float h = Hw[lane * 16 + c];
    rgb_out[p * 3 + c] = 1.f / (1.f + __expf(-h));
  }
}

// ===========================================================================
// Kernel 2: per-ray volume rendering. One wave32 per ray; coalesced chunked
// loads + __shfl_up inclusive scan for exact exclusive transmittance.
// ===========================================================================
__global__ __launch_bounds__(128) void render_kernel(
    const float* __restrict__ rays_o, const float* __restrict__ rays_d,
    const float* __restrict__ bg,
    const float* __restrict__ sigma, const float* __restrict__ rgb,
    float* __restrict__ out)
{
  const int ray  = blockIdx.x * 4 + (threadIdx.x >> 5);
  const int lane = threadIdx.x & 31;

  float ox = rays_o[ray * 3 + 0], oy = rays_o[ray * 3 + 1], oz = rays_o[ray * 3 + 2];
  float dx = rays_d[ray * 3 + 0], dy = rays_d[ray * 3 + 1], dz = rays_d[ray * 3 + 2];
  float rn = __frsqrt_rn(dx * dx + dy * dy + dz * dz);
  dx *= rn; dy *= rn; dz *= rn;
  float bb = ox * dx + oy * dy + oz * dz;
  float cc = ox * ox + oy * oy + oz * oz - RADIUS_F * RADIUS_F;
  float disc = __fsqrt_rn(fmaxf(bb * bb - cc, 0.f));
  float tn = fmaxf(-bb - disc, 0.f);
  float tf = fmaxf(-bb + disc, tn);
  float delta = (tf - tn) * (1.f / 256.f);

  float carry = 0.f, ar = 0.f, ag = 0.f, ab = 0.f, alpha = 0.f;
#pragma unroll
  for (int chunk = 0; chunk < 8; ++chunk) {
    int pidx = ray * 256 + chunk * 32 + lane;
    float tau = sigma[pidx] * delta;
    float scan = tau;
#pragma unroll
    for (int off = 1; off < 32; off <<= 1) {
      float v = __shfl_up(scan, off, 32);
      if (lane >= off) scan += v;
    }
    float excl = carry + scan - tau;
    float T = __expf(-excl);
    float w = T * (1.f - __expf(-tau));
    ar    += w * rgb[pidx * 3 + 0];
    ag    += w * rgb[pidx * 3 + 1];
    ab    += w * rgb[pidx * 3 + 2];
    alpha += w;
    carry += __shfl(scan, 31, 32);
  }
#pragma unroll
  for (int off = 16; off > 0; off >>= 1) {
    ar    += __shfl_down(ar,    off, 32);
    ag    += __shfl_down(ag,    off, 32);
    ab    += __shfl_down(ab,    off, 32);
    alpha += __shfl_down(alpha, off, 32);
  }
  if (lane == 0) {
    out[ray * 3 + 0] = ar + (1.f - alpha) * bg[ray * 3 + 0];
    out[ray * 3 + 1] = ag + (1.f - alpha) * bg[ray * 3 + 1];
    out[ray * 3 + 2] = ab + (1.f - alpha) * bg[ray * 3 + 2];
  }
}

// ===========================================================================
extern "C" void kernel_launch(void* const* d_in, const int* in_sizes, int n_in,
                              void* d_out, int out_size, void* d_ws, size_t ws_size,
                              hipStream_t stream) {
  const float* rays_o = (const float*)d_in[0];
  const float* rays_d = (const float*)d_in[1];
  const float* bg     = (const float*)d_in[2];
  const float* plane  = (const float*)d_in[3];
  const float* feat   = (const float*)d_in[4];
  const float* W1  = (const float*)d_in[5];
  const float* b1  = (const float*)d_in[6];
  const float* W2  = (const float*)d_in[7];
  const float* b2  = (const float*)d_in[8];
  const float* Wc1 = (const float*)d_in[9];
  const float* bc1 = (const float*)d_in[10];
  const float* Wc2 = (const float*)d_in[11];
  const float* bc2 = (const float*)d_in[12];

  float* sigma = (float*)d_ws;          // NPTS floats
  float* rgb   = sigma + NPTS;          // 3*NPTS floats
  float* out   = (float*)d_out;         // 4096 x 3

  point_kernel<<<NPTS / 128, 128, 0, stream>>>(
      rays_o, rays_d, plane, feat, W1, b1, W2, b2, Wc1, bc1, Wc2, bc2,
      sigma, rgb);
  render_kernel<<<NRAYS / 4, 128, 0, stream>>>(
      rays_o, rays_d, bg, sigma, rgb, out);
}
